// MultiscaleBlock_50594714746912
// MI455X (gfx1250) — matrix operand order repaired and checked by reading, not compiled, not run on hardware
//
#include <hip/hip_runtime.h>
#include <hip/hip_bf16.h>

// ---------------------------------------------------------------------------
// Problem constants
// ---------------------------------------------------------------------------
static constexpr int NB   = 64;          // batch
static constexpr int NS   = 257;         // sequence
static constexpr int NC   = 768;         // channels
static constexpr int NH   = 12;          // heads
static constexpr int ND   = 64;          // head dim
static constexpr int NHID = 3072;        // mlp hidden
static constexpr int NT   = NB * NS;     // tokens = 16448
static constexpr float ATTN_SCALE = 0.125f;   // 64^-0.5

typedef __attribute__((ext_vector_type(16))) __bf16 v16bf;
typedef __attribute__((ext_vector_type(8)))  float  v8f;

union BfFrag { v16bf v; uint4 q[2]; };

// ---- gfx1250 async global->LDS path (guarded; fallback = load + ds_store) --
#if defined(__has_builtin)
#  if __has_builtin(__builtin_amdgcn_global_load_async_to_lds_b128) && \
      __has_builtin(__builtin_amdgcn_s_wait_asynccnt)
#    define ASYNC_LDS 1
#  endif
#endif
#ifndef ASYNC_LDS
#  define ASYNC_LDS 0
#endif

#if ASYNC_LDS
// Builtin signature (from hipcc diagnostic): param0 is
//   'int __attribute__((vector_size(16))) __device__ *'  (global int4*)
typedef int i4 __attribute__((vector_size(4 * sizeof(int))));
typedef __attribute__((address_space(1))) i4* gas_i4;
typedef __attribute__((address_space(3))) i4* las_i4;
__device__ __forceinline__ void async_cp16(const void* g, void* l) {
    __builtin_amdgcn_global_load_async_to_lds_b128((gas_i4)g, (las_i4)l, 0, 0);
}
#endif

// ---------------------------------------------------------------------------
// Helpers
// ---------------------------------------------------------------------------
__device__ __forceinline__ unsigned short f2bf(float f) {
    union { float f; unsigned u; } v; v.f = f;
    unsigned r = v.u + 0x7FFFu + ((v.u >> 16) & 1u);   // round-nearest-even
    return (unsigned short)(r >> 16);
}

__device__ __forceinline__ float gelu_exact(float x) {
    return 0.5f * x * (1.0f + erff(x * 0.70710678118654752f));
}

__device__ __forceinline__ float arf_act(float x) {
    float ex  = __expf(x);
    float emx = __expf(-x);
    float t = (ex - emx) / (ex + __expf(-x - 4.0f));
    return t < 0.0f ? 0.0f : t;
}

// ---------------------------------------------------------------------------
// GEMM:  out[M,Nout] = act( A_bf16[M,K] * W_bf16[Nout,K]^T + bias ) (+residual)
// Block tile 128x128, K-step 32, double-buffered LDS (async global->LDS when
// available).  8 waves; each wave computes 32x64 via a 2x4 grid of
// v_wmma_f32_16x16x32_bf16 -> 8 WMMA per K-step per wave.
// act: 0 = none, 1 = gelu(exact), 2 = arf
// ---------------------------------------------------------------------------
__global__ __launch_bounds__(256)
void gemm_bf16_wmma(const unsigned short* __restrict__ A,
                    const unsigned short* __restrict__ W,
                    const float* bias,
                    const float* residual,       // may alias out
                    float* outF,                 // one of outF / outH non-null
                    unsigned short* outH,
                    int M, int K, int Nout, int act)
{
    __shared__ __attribute__((aligned(16))) unsigned short sA[2][128][40];
    __shared__ __attribute__((aligned(16))) unsigned short sW[2][128][40];

    const int tid  = threadIdx.x;
    const int lane = tid & 31;
    const int wid  = tid >> 5;
    const int wm   = wid >> 1;          // 0..3  -> 32-row slice
    const int wn   = wid & 1;           // 0..1  -> 64-col slice
    const int m0   = blockIdx.y * 128;
    const int n0   = blockIdx.x * 128;

    const int hr = lane >> 4;           // half-wave
    const int lr = lane & 15;           // row/col inside 16

    v8f acc[2][4];
#pragma unroll
    for (int mi = 0; mi < 2; ++mi)
#pragma unroll
        for (int ni = 0; ni < 4; ++ni)
#pragma unroll
            for (int v = 0; v < 8; ++v)
                acc[mi][ni][v] = 0.0f;

    // stage one 128x32 A tile + 128x32 W tile into LDS buffer `buf`
    auto stage = [&](int buf, int k0) {
#pragma unroll
        for (int i = 0; i < 2; ++i) {
            int q  = tid * 2 + i;
            int r  = q >> 2, cq = q & 3;
            int grow = m0 + r;
            unsigned short* ldst = &sA[buf][r][cq * 8];
            if (grow < M) {
#if ASYNC_LDS
                async_cp16(A + (size_t)grow * K + k0 + cq * 8, ldst);
#else
                *(uint4*)ldst = *(const uint4*)(A + (size_t)grow * K + k0 + cq * 8);
#endif
            } else {
                *(uint4*)ldst = make_uint4(0u, 0u, 0u, 0u);
            }
        }
#pragma unroll
        for (int i = 0; i < 2; ++i) {
            int q  = tid * 2 + i;
            int r  = q >> 2, cq = q & 3;
            unsigned short* ldst = &sW[buf][r][cq * 8];
#if ASYNC_LDS
            async_cp16(W + (size_t)(n0 + r) * K + k0 + cq * 8, ldst);
#else
            *(uint4*)ldst = *(const uint4*)(W + (size_t)(n0 + r) * K + k0 + cq * 8);
#endif
        }
    };

    stage(0, 0);
#if ASYNC_LDS
    __builtin_amdgcn_s_wait_asynccnt(0);
#endif
    __syncthreads();

    int cur = 0;
    for (int k0 = 0; k0 < K; k0 += 32) {
        // overlap: fetch next K tile into the other buffer
        if (k0 + 32 < K) stage(cur ^ 1, k0 + 32);

        // ---- fragments from current buffer (ISA VGPR layouts) ----
        // A 16x32: lanes 0-15 rows, VGPR0-3 K=h*8.., VGPR4-7 K=16+h*8..
        BfFrag a[2], bm[4];
#pragma unroll
        for (int mi = 0; mi < 2; ++mi) {
            int ar = wm * 32 + mi * 16 + lr;
            a[mi].q[0] = *(const uint4*)&sA[cur][ar][hr * 8];
            a[mi].q[1] = *(const uint4*)&sA[cur][ar][16 + hr * 8];
        }
        // B 32x16: lane = h*16+n holds W[n][k0 + h*8 ..+7] and W[n][k0 + 16 + h*8 ..+7]
#pragma unroll
        for (int ni = 0; ni < 4; ++ni) {
            int br = wn * 64 + ni * 16 + lr;
            bm[ni].q[0] = *(const uint4*)&sW[cur][br][hr * 8];
            bm[ni].q[1] = *(const uint4*)&sW[cur][br][16 + hr * 8];
        }

#pragma unroll
        for (int mi = 0; mi < 2; ++mi)
#pragma unroll
            for (int ni = 0; ni < 4; ++ni)
                acc[mi][ni] = __builtin_amdgcn_wmma_f32_16x16x32_bf16(
                    false, a[mi].v, false, bm[ni].v,
                    (short)0, acc[mi][ni], false, false);

#if ASYNC_LDS
        __builtin_amdgcn_s_wait_asynccnt(0);
#endif
        __syncthreads();
        cur ^= 1;
    }

    // ---- epilogue: D layout: VGPR v -> row v + hr*8, col lr ----
#pragma unroll
    for (int mi = 0; mi < 2; ++mi) {
#pragma unroll
        for (int ni = 0; ni < 4; ++ni) {
#pragma unroll
            for (int v = 0; v < 8; ++v) {
                int m = m0 + wm * 32 + mi * 16 + hr * 8 + v;
                int n = n0 + wn * 64 + ni * 16 + lr;
                if (m < M) {
                    float val = acc[mi][ni][v];
                    if (bias) val += bias[n];
                    if (act == 1)      val = gelu_exact(val);
                    else if (act == 2) val = arf_act(val);
                    if (residual) val += residual[(size_t)m * Nout + n];
                    if (outF) outF[(size_t)m * Nout + n] = val;
                    else      outH[(size_t)m * Nout + n] = f2bf(val);
                }
            }
        }
    }
}

// ---------------------------------------------------------------------------
// Attention: per thread = one (b,h,query). Multi-pass softmax with K/V tiles
// staged in LDS. Writes O as bf16 in (b,n,h,d)->[T,C] layout.
// If attn_sum != nullptr, also accumulates head-summed probabilities
// (cross-attention path) with global float atomics.
// ---------------------------------------------------------------------------
__global__ __launch_bounds__(128)
void attention_kernel(const float* __restrict__ qkvQ,
                      const float* __restrict__ qkvKV,
                      unsigned short* __restrict__ o_out,
                      float* attn_sum)
{
    __shared__ float sK[64][ND + 1];
    __shared__ float sV[64][ND + 1];

    const int tid   = threadIdx.x;
    const int chunk = blockIdx.x;          // 0..2
    const int h     = blockIdx.y;          // 0..11
    const int b     = blockIdx.z;          // 0..63
    const int n     = chunk * 128 + tid;
    const bool active = (n < NS);

    float q[ND];
    if (active) {
        const float* qp = qkvQ + (size_t)(b * NS + n) * (3 * NC) + h * ND;
#pragma unroll
        for (int d = 0; d < ND; ++d) q[d] = qp[d] * ATTN_SCALE;
    } else {
#pragma unroll
        for (int d = 0; d < ND; ++d) q[d] = 0.0f;
    }

    // pass 1: row max
    float mx = -1e30f;
    for (int kt = 0; kt < NS; kt += 64) {
        for (int i = tid; i < 64 * ND; i += 128) {
            int r = i >> 6, c = i & 63;
            int kg = kt + r;
            sK[r][c] = (kg < NS)
                ? qkvKV[(size_t)(b * NS + kg) * (3 * NC) + NC + h * ND + c]
                : 0.0f;
        }
        __syncthreads();
        if (active) {
            int kmax = NS - kt; if (kmax > 64) kmax = 64;
            for (int kk = 0; kk < kmax; ++kk) {
                float s = 0.0f;
#pragma unroll
                for (int d = 0; d < ND; ++d) s += q[d] * sK[kk][d];
                mx = fmaxf(mx, s);
            }
        }
        __syncthreads();
    }

    // pass 2: exp-sum + weighted V accumulation
    float l = 0.0f;
    float o[ND];
#pragma unroll
    for (int d = 0; d < ND; ++d) o[d] = 0.0f;

    for (int kt = 0; kt < NS; kt += 64) {
        for (int i = tid; i < 64 * ND; i += 128) {
            int r = i >> 6, c = i & 63;
            int kg = kt + r;
            size_t base = (size_t)(b * NS + kg) * (3 * NC) + h * ND + c;
            sK[r][c] = (kg < NS) ? qkvKV[base + NC]     : 0.0f;
            sV[r][c] = (kg < NS) ? qkvKV[base + 2 * NC] : 0.0f;
        }
        __syncthreads();
        if (active) {
            int kmax = NS - kt; if (kmax > 64) kmax = 64;
            for (int kk = 0; kk < kmax; ++kk) {
                float s = 0.0f;
#pragma unroll
                for (int d = 0; d < ND; ++d) s += q[d] * sK[kk][d];
                float e = __expf(s - mx);
                l += e;
#pragma unroll
                for (int d = 0; d < ND; ++d) o[d] += e * sV[kk][d];
            }
        }
        __syncthreads();
    }

    float inv = active ? (1.0f / l) : 0.0f;
    if (active) {
        unsigned short* op = o_out + (size_t)(b * NS + n) * NC + h * ND;
#pragma unroll
        for (int d = 0; d < ND; ++d) op[d] = f2bf(o[d] * inv);
    }

    // pass 3 (cross only): head-summed normalized probabilities
    if (attn_sum) {
        for (int kt = 0; kt < NS; kt += 64) {
            for (int i = tid; i < 64 * ND; i += 128) {
                int r = i >> 6, c = i & 63;
                int kg = kt + r;
                sK[r][c] = (kg < NS)
                    ? qkvKV[(size_t)(b * NS + kg) * (3 * NC) + NC + h * ND + c]
                    : 0.0f;
            }
            __syncthreads();
            if (active) {
                int kmax = NS - kt; if (kmax > 64) kmax = 64;
                for (int kk = 0; kk < kmax; ++kk) {
                    float s = 0.0f;
#pragma unroll
                    for (int d = 0; d < ND; ++d) s += q[d] * sK[kk][d];
                    float p = __expf(s - mx) * inv;
                    atomicAdd(&attn_sum[((size_t)b * NS + n) * NS + kt + kk], p);
                }
            }
            __syncthreads();
        }
    }
}

// ---------------------------------------------------------------------------
// LayerNorm over C=768, one block per token row, bf16 output.
// ---------------------------------------------------------------------------
__global__ __launch_bounds__(256)
void layernorm_bf16(const float* __restrict__ x,
                    const float* __restrict__ g,
                    const float* __restrict__ beta,
                    unsigned short* __restrict__ out)
{
    __shared__ float red[256];
    const int row = blockIdx.x;
    const int tid = threadIdx.x;
    const float* xr = x + (size_t)row * NC;

    float v0 = xr[tid], v1 = xr[tid + 256], v2 = xr[tid + 512];
    red[tid] = v0 + v1 + v2;
    __syncthreads();
    for (int st = 128; st > 0; st >>= 1) {
        if (tid < st) red[tid] += red[tid + st];
        __syncthreads();
    }
    float mean = red[0] * (1.0f / (float)NC);
    __syncthreads();

    float d0 = v0 - mean, d1 = v1 - mean, d2 = v2 - mean;
    red[tid] = d0 * d0 + d1 * d1 + d2 * d2;
    __syncthreads();
    for (int st = 128; st > 0; st >>= 1) {
        if (tid < st) red[tid] += red[tid + st];
        __syncthreads();
    }
    float rstd = rsqrtf(red[0] * (1.0f / (float)NC) + 1e-5f);
    __syncthreads();

    unsigned short* orow = out + (size_t)row * NC;
    orow[tid]       = f2bf(d0 * rstd * g[tid]       + beta[tid]);
    orow[tid + 256] = f2bf(d1 * rstd * g[tid + 256] + beta[tid + 256]);
    orow[tid + 512] = f2bf(d2 * rstd * g[tid + 512] + beta[tid + 512]);
}

// ---------------------------------------------------------------------------
// Elementwise kernels
// ---------------------------------------------------------------------------
__global__ void conv_f32_bf16(const float* __restrict__ x,
                              unsigned short* __restrict__ y, int n)
{
    int i = blockIdx.x * blockDim.x + threadIdx.x;
    if (i < n) y[i] = f2bf(x[i]);
}

__global__ void add_pos_kernel(const float* __restrict__ src,
                               const float* __restrict__ pos,
                               float* __restrict__ out, int n)
{
    int i = blockIdx.x * blockDim.x + threadIdx.x;
    if (i < n) out[i] = src[i] + pos[i % (NS * NC)];
}

__global__ void axpy_kernel(float* __restrict__ y,
                            const float* __restrict__ x, int n)
{
    int i = blockIdx.x * blockDim.x + threadIdx.x;
    if (i < n) y[i] += x[i];
}

// tgt_out[b,j,c] = tgt[b,j,c] * (1 + sa_w[c] * sigmoid(mean_n attn_sum[b,n,j]))
__global__ __launch_bounds__(256)
void gate_kernel(const float* __restrict__ attn_sum,
                 const float* __restrict__ tgt_acc,
                 const float* __restrict__ sa_w,
                 float* __restrict__ out)
{
    __shared__ float red[256];
    const int j = blockIdx.x;      // 0..256
    const int b = blockIdx.y;      // 0..63
    const int tid = threadIdx.x;

    float s = 0.0f;
    for (int n = tid; n < NS; n += 256)
        s += attn_sum[((size_t)b * NS + n) * NS + j];
    red[tid] = s;
    __syncthreads();
    for (int st = 128; st > 0; st >>= 1) {
        if (tid < st) red[tid] += red[tid + st];
        __syncthreads();
    }
    float heat = red[0] * (1.0f / (float)NS);
    float gate = 1.0f / (1.0f + __expf(-heat));
    __syncthreads();

    const float* tr = tgt_acc + (size_t)(b * NS + j) * NC;
    float* orow     = out     + (size_t)(b * NS + j) * NC;
    for (int c = tid; c < NC; c += 256)
        orow[c] = tr[c] * (1.0f + sa_w[c] * gate);
}

__global__ void attm_copy_kernel(const float* __restrict__ attn_sum,
                                 float* __restrict__ out)
{
    int idx = blockIdx.x * 256 + threadIdx.x;
    if (idx < NB * 256 * 256) {
        int j = idx & 255;
        int i = (idx >> 8) & 255;
        int b = idx >> 16;
        out[idx] = attn_sum[((size_t)b * NS + (i + 1)) * NS + (j + 1)];
    }
}

// ---------------------------------------------------------------------------
// Host orchestration
// ---------------------------------------------------------------------------
extern "C" void kernel_launch(void* const* d_in, const int* in_sizes, int n_in,
                              void* d_out, int out_size, void* d_ws, size_t ws_size,
                              hipStream_t stream)
{
    (void)in_sizes; (void)n_in; (void)out_size; (void)ws_size;

    const float* src          = (const float*)d_in[0];
    const float* tgt          = (const float*)d_in[1];
    const float* sattn_qkv_w  = (const float*)d_in[2];
    const float* sattn_proj_w = (const float*)d_in[3];
    const float* sattn_proj_b = (const float*)d_in[4];
    const float* cattn_qkv_w  = (const float*)d_in[5];
    const float* cattn_proj_w = (const float*)d_in[6];
    const float* cattn_proj_b = (const float*)d_in[7];
    const float* norm1_g      = (const float*)d_in[8];
    const float* norm1_b      = (const float*)d_in[9];
    const float* norm2_g      = (const float*)d_in[10];
    const float* norm2_b      = (const float*)d_in[11];
    const float* fc1_w        = (const float*)d_in[12];
    const float* fc1_b        = (const float*)d_in[13];
    const float* fc2_w        = (const float*)d_in[14];
    const float* fc2_b        = (const float*)d_in[15];
    const float* sa_weight    = (const float*)d_in[16];
    const float* pos_embed    = (const float*)d_in[17];

    float* out_src  = (float*)d_out;                       // [T, C]
    float* out_tgt  = out_src + (size_t)NT * NC;           // [T, C]
    float* out_attm = out_tgt + (size_t)NT * NC;           // [B, 256, 256]

    // ---- workspace bump allocator (256B aligned) ----
    char* ws = (char*)d_ws;
    size_t off = 0;
    auto alloc = [&](size_t bytes) -> void* {
        off = (off + 255) & ~(size_t)255;
        void* p = ws + off;
        off += bytes;
        return p;
    };

    float* src_acc  = (float*)alloc(sizeof(float) * NT * NC);
    float* tgt_acc  = (float*)alloc(sizeof(float) * NT * NC);
    float* t_tmp    = (float*)alloc(sizeof(float) * NT * NC);
    unsigned short* ln_bf  = (unsigned short*)alloc(sizeof(unsigned short) * NT * NC);
    unsigned short* o_bf   = (unsigned short*)alloc(sizeof(unsigned short) * NT * NC);
    unsigned short* h_bf   = (unsigned short*)alloc(sizeof(unsigned short) * (size_t)NT * NHID);
    float* qkvA     = (float*)alloc(sizeof(float) * (size_t)NT * 3 * NC);
    float* qkvB     = (float*)alloc(sizeof(float) * (size_t)NT * 3 * NC);
    float* attn_sum = (float*)alloc(sizeof(float) * (size_t)NB * NS * NS);

    unsigned short* wqs = (unsigned short*)alloc(sizeof(unsigned short) * 3 * NC * NC);
    unsigned short* wps = (unsigned short*)alloc(sizeof(unsigned short) * NC * NC);
    unsigned short* wqc = (unsigned short*)alloc(sizeof(unsigned short) * 3 * NC * NC);
    unsigned short* wpc = (unsigned short*)alloc(sizeof(unsigned short) * NC * NC);
    unsigned short* w1  = (unsigned short*)alloc(sizeof(unsigned short) * (size_t)NHID * NC);
    unsigned short* w2  = (unsigned short*)alloc(sizeof(unsigned short) * (size_t)NC * NHID);

    auto conv = [&](const float* s, unsigned short* d, int n) {
        conv_f32_bf16<<<(n + 255) / 256, 256, 0, stream>>>(s, d, n);
    };
    auto gemm = [&](const unsigned short* A, const unsigned short* Wm,
                    const float* bias, const float* resid,
                    float* oF, unsigned short* oH, int M, int K, int Nout, int act) {
        dim3 grid(Nout / 128, (M + 127) / 128);
        gemm_bf16_wmma<<<grid, 256, 0, stream>>>(A, Wm, bias, resid, oF, oH, M, K, Nout, act);
    };
    auto layernorm = [&](const float* x, const float* g, const float* b,
                         unsigned short* o) {
        layernorm_bf16<<<NT, 256, 0, stream>>>(x, g, b, o);
    };
    const int nTok = NT * NC;

    // ---- weight conversion to bf16 ----
    conv(sattn_qkv_w,  wqs, 3 * NC * NC);
    conv(sattn_proj_w, wps, NC * NC);
    conv(cattn_qkv_w,  wqc, 3 * NC * NC);
    conv(cattn_proj_w, wpc, NC * NC);
    conv(fc1_w,        w1,  NHID * NC);
    conv(fc2_w,        w2,  NC * NHID);

    // ---- src = src + pos_embed ; tgt_acc = tgt ----
    add_pos_kernel<<<(nTok + 255) / 256, 256, 0, stream>>>(src, pos_embed, src_acc, nTok);
    (void)hipMemcpyAsync(tgt_acc, tgt, sizeof(float) * nTok, hipMemcpyDeviceToDevice, stream);

    dim3 attn_grid(3, NH, NB);   // ceil(257/128) query chunks, heads, batch

    // ================= self-attention + MLP : src =================
    layernorm(src_acc, norm1_g, norm1_b, ln_bf);
    gemm(ln_bf, wqs, nullptr, nullptr, qkvA, nullptr, NT, NC, 3 * NC, 0);
    attention_kernel<<<attn_grid, 128, 0, stream>>>(qkvA, qkvA, o_bf, nullptr);
    gemm(o_bf, wps, sattn_proj_b, nullptr, t_tmp, nullptr, NT, NC, NC, 0);     // srct
    axpy_kernel<<<(nTok + 255) / 256, 256, 0, stream>>>(src_acc, t_tmp, nTok); // src += srct
    layernorm(t_tmp, norm2_g, norm2_b, ln_bf);
    gemm(ln_bf, w1, fc1_b, nullptr, nullptr, h_bf, NT, NC, NHID, 1);           // gelu
    gemm(h_bf, w2, fc2_b, src_acc, src_acc, nullptr, NT, NHID, NC, 0);         // src += mlp

    // ================= self-attention + MLP : tgt =================
    layernorm(tgt_acc, norm1_g, norm1_b, ln_bf);
    gemm(ln_bf, wqs, nullptr, nullptr, qkvA, nullptr, NT, NC, 3 * NC, 0);
    attention_kernel<<<attn_grid, 128, 0, stream>>>(qkvA, qkvA, o_bf, nullptr);
    gemm(o_bf, wps, sattn_proj_b, nullptr, t_tmp, nullptr, NT, NC, NC, 0);     // tgtt
    axpy_kernel<<<(nTok + 255) / 256, 256, 0, stream>>>(tgt_acc, t_tmp, nTok); // tgt += tgtt
    layernorm(t_tmp, norm2_g, norm2_b, ln_bf);
    gemm(ln_bf, w1, fc1_b, nullptr, nullptr, h_bf, NT, NC, NHID, 1);
    gemm(h_bf, w2, fc2_b, tgt_acc, tgt_acc, nullptr, NT, NHID, NC, 0);         // tgt += mlp

    // ================= cross-attention =================
    (void)hipMemsetAsync(attn_sum, 0, sizeof(float) * (size_t)NB * NS * NS, stream);
    layernorm(src_acc, norm1_g, norm1_b, ln_bf);
    gemm(ln_bf, wqc, nullptr, nullptr, qkvA, nullptr, NT, NC, 3 * NC, 0);      // q from src
    layernorm(tgt_acc, norm1_g, norm1_b, ln_bf);
    gemm(ln_bf, wqc, nullptr, nullptr, qkvB, nullptr, NT, NC, 3 * NC, 0);      // k,v from tgt
    attention_kernel<<<attn_grid, 128, 0, stream>>>(qkvA, qkvB, o_bf, attn_sum);
    gemm(o_bf, wpc, cattn_proj_b, nullptr, t_tmp, nullptr, NT, NC, NC, 2);     // arf(proj+b)
    axpy_kernel<<<(nTok + 255) / 256, 256, 0, stream>>>(src_acc, t_tmp, nTok); // src += srct
    layernorm(t_tmp, norm2_g, norm2_b, ln_bf);
    gemm(ln_bf, w1, fc1_b, nullptr, nullptr, h_bf, NT, NC, NHID, 1);
    gemm(h_bf, w2, fc2_b, src_acc, out_src, nullptr, NT, NHID, NC, 0);         // final src

    // ================= gating + attm extraction =================
    dim3 gate_grid(NS, NB);
    gate_kernel<<<gate_grid, 256, 0, stream>>>(attn_sum, tgt_acc, sa_weight, out_tgt);
    attm_copy_kernel<<<(NB * 256 * 256 + 255) / 256, 256, 0, stream>>>(attn_sum, out_attm);
}
